// LocalEncoder_hivt_50697793962571
// MI455X (gfx1250) — matrix operand order repaired and checked by reading, not compiled, not run on hardware
//
#include <hip/hip_runtime.h>
#include <hip/hip_bf16.h>

// CDNA5 / gfx1250 implementation of HiVT LocalEncoder layer.
// All GEMMs via v_wmma_f32_16x16x32_f16 (f16 A/B, f32 accum).
// Weights pre-packed to f16 in WMMA B-fragment order (no per-wave cvt).
// LayerNorm parallelized across all 32 lanes via __shfl_xor(16) (wave32).

typedef __attribute__((ext_vector_type(16))) _Float16 v16h;
typedef __attribute__((ext_vector_type(8)))  float    v8f;

#define DEV static __device__ __forceinline__

DEV int lane_id() { return (int)(threadIdx.x & 31u); }
DEV int imin(int a, int b) { return a < b ? a : b; }

// ---- ordered-uint encoding for float atomicMax ----
DEV unsigned int f2ord(float f) {
  unsigned int u = __float_as_uint(f);
  return (u & 0x80000000u) ? ~u : (u | 0x80000000u);
}
DEV float ord2f(unsigned int k) {
  unsigned int u = (k & 0x80000000u) ? (k ^ 0x80000000u) : ~k;
  return __uint_as_float(u);
}

// ---- WMMA fragment loaders (per CDNA5 ISA layouts, wave32) ----
// A: 16x32 f16 tile from LDS (row-major, leading dim lda)
DEV v16h load_a(const _Float16* A, int lda) {
  int lane = lane_id();
  int row  = lane & 15;
  int kb   = (lane >> 4) << 3;   // 0 or 8
  v16h a;
#pragma unroll
  for (int g = 0; g < 2; ++g)
#pragma unroll
    for (int r = 0; r < 4; ++r) {
      int k = g * 16 + kb + 2 * r;
      a[g * 8 + 2 * r]     = A[row * lda + k];
      a[g * 8 + 2 * r + 1] = A[row * lda + k + 1];
    }
  return a;
}

// B: pre-packed f16 fragment; each lane's 16 halves are contiguous (32B aligned)
DEV v16h load_bp(const _Float16* __restrict__ P, int kc, int k0, int n0) {
  int lane = lane_id();
  size_t off = ((size_t)((n0 >> 4) * kc + (k0 >> 5)) << 9) + ((size_t)lane << 4);
  return *(const v16h*)(P + off);
}

// 16x16 output tile accumulated over K (multiple of 32), packed weights
DEV v8f wmma_tile(const _Float16* A, int lda, const _Float16* Wp, int kc,
                  int n0, int K) {
  v8f acc = {};
  for (int k0 = 0; k0 < K; k0 += 32) {
    v16h a = load_a(A + k0, lda);
    v16h b = load_bp(Wp, kc, k0, n0);
    acc = __builtin_amdgcn_wmma_f32_16x16x32_f16(false, a, false, b,
                                                 (short)0, acc, false, false);
  }
  return acc;
}

// D-layout store: lanes 0-15 -> N=lane, rows M=0..7; lanes 16-31 -> M=8..15
DEV void store_f32(float* out, int ldo, int n0, const float* bias, v8f d) {
  int lane = lane_id();
  int col  = n0 + (lane & 15);
  int m0   = (lane >> 4) << 3;
  float bv = bias[col];
#pragma unroll
  for (int r = 0; r < 8; ++r) out[(m0 + r) * ldo + col] = d[r] + bv;
}

DEV void store_f32_acc(float* out, int ldo, int n0, const float* bias, v8f d) {
  int lane = lane_id();
  int col  = n0 + (lane & 15);
  int m0   = (lane >> 4) << 3;
  float bv = bias[col];
#pragma unroll
  for (int r = 0; r < 8; ++r) out[(m0 + r) * ldo + col] += d[r] + bv;
}

DEV void store_h16_relu(_Float16* out, int ldo, int n0, const float* bias, v8f d) {
  int lane = lane_id();
  int col  = n0 + (lane & 15);
  int m0   = (lane >> 4) << 3;
  float bv = bias[col];
#pragma unroll
  for (int r = 0; r < 8; ++r) {
    float y = fmaxf(d[r] + bv, 0.f);
    out[(m0 + r) * ldo + col] = (_Float16)y;
  }
}

// full 16xNout GEMM: out = A(16xK,f16 in LDS) @ Wp(packed f16) + bias
DEV void gemm_f32(const _Float16* A, int lda, const _Float16* Wp, const float* bias,
                  int K, int Nout, float* out) {
  int kc = K >> 5;
  for (int n0 = 0; n0 < Nout; n0 += 16)
    store_f32(out, Nout, n0, bias, wmma_tile(A, lda, Wp, kc, n0, K));
}
// accumulate version: out += A@Wp + bias
DEV void gemm_f32_acc(const _Float16* A, int lda, const _Float16* Wp, const float* bias,
                      int K, int Nout, float* out) {
  int kc = K >> 5;
  for (int n0 = 0; n0 < Nout; n0 += 16)
    store_f32_acc(out, Nout, n0, bias, wmma_tile(A, lda, Wp, kc, n0, K));
}

// LayerNorm over a 16x64 f32 LDS tile. All 32 lanes: lane handles
// row = lane&15, cols [(lane>>4)*32, +32); halves combined via shfl_xor(16).
// Optional ReLU; optional f32 and/or f16 row-major outputs (only rows < nvalid).
DEV void ln_tile(const float* tile, int ld, const float* g, const float* b,
                 bool relu, float* outf, int ldf, _Float16* outh, int ldh,
                 int nvalid) {
  int lane = lane_id();
  int row = lane & 15;
  int c0 = (lane >> 4) << 5;
  const float* in = tile + row * ld;
  float s = 0.f, s2 = 0.f;
#pragma unroll
  for (int c = 0; c < 32; ++c) { float v = in[c0 + c]; s += v; s2 += v * v; }
  s  += __shfl_xor(s, 16, 32);
  s2 += __shfl_xor(s2, 16, 32);
  float m   = s * (1.f / 64.f);
  float var = fmaxf(s2 * (1.f / 64.f) - m * m, 0.f);
  float inv = rsqrtf(var + 1e-5f);
  bool wr = row < nvalid;
#pragma unroll
  for (int c = 0; c < 32; ++c) {
    int cc = c0 + c;
    float y = (in[cc] - m) * inv * g[cc] + b[cc];
    if (relu) y = fmaxf(y, 0.f);
    if (outf && wr) outf[row * ldf + cc] = y;
    if (outh && wr) outh[row * ldh + cc] = (_Float16)y;
  }
}

// ===================== kernels =====================

// Pack fp32 weight W[K][N] into f16 WMMA B-fragment order.
__global__ void k_pack(const float* __restrict__ W, _Float16* __restrict__ P,
                       int K, int N) {
  int idx = blockIdx.x * blockDim.x + threadIdx.x;
  if (idx >= K * N) return;
  int kc = K >> 5;
  int tile   = idx >> 9;
  int within = idx & 511;
  int lane = within >> 4;
  int j    = within & 15;
  int n0 = (tile / kc) << 4;
  int k0 = (tile % kc) << 5;
  int col  = n0 + (lane & 15);
  int krow = k0 + ((lane >> 4) << 3) + (j & 7) + ((j >> 3) << 4);
  P[idx] = (_Float16)W[(size_t)krow * N + col];
}

__global__ void k_init(float* agg, float* den, unsigned int* amax, int N) {
  int i = blockIdx.x * blockDim.x + threadIdx.x;
  if (i < N * 64) agg[i] = 0.f;
  if (i < N * 8) { den[i] = 0.f; amax[i] = 0u; }  // 0 == ordered(-inf-ish)
}

// ce = single_embed(rot(x)); bos replace; ce_n = LN(ce,norm1); me_n = LN(me,norm3)
__global__ __launch_bounds__(32) void k_node_embed(
    const float* __restrict__ x, const int* __restrict__ tptr,
    const unsigned char* __restrict__ bos_mask,
    const float* __restrict__ motion, const float* __restrict__ rot,
    const float* W1, const float* b1,
    const _Float16* __restrict__ pW2, const float* b2,
    const _Float16* __restrict__ pW3, const float* b3,
    const float* n1b, const float* n1g, const float* n2b, const float* n2g,
    const float* n3b, const float* n3g,
    const float* norm1b, const float* norm1g,
    const float* norm3b, const float* norm3g,
    const float* bos_token,
    float* ce_out, _Float16* cenh, _Float16* menh, int N) {
  __shared__ float    fA[16 * 64];
  __shared__ float    fB[16 * 64];
  __shared__ _Float16 hX[16 * 64];
  int lane = lane_id();
  int row0 = blockIdx.x * 16;
  int t = tptr[0];
  int r  = lane & 15;
  int cb = (lane >> 4) << 5;

  {
    int row = imin(row0 + r, N - 1);
    float x0 = x[row * 2], x1 = x[row * 2 + 1];
    const float* R = rot + (size_t)row * 4;
    float xr0 = x0 * R[0] + x1 * R[2];
    float xr1 = x0 * R[1] + x1 * R[3];
    for (int c = cb; c < cb + 32; ++c)
      fA[r * 64 + c] = xr0 * W1[c] + xr1 * W1[64 + c] + b1[c];
  }
  __syncthreads();
  ln_tile(fA, 64, n1g, n1b, true, nullptr, 0, hX, 64, 16);
  __syncthreads();
  gemm_f32(hX, 64, pW2, b2, 64, 64, fB);
  __syncthreads();
  ln_tile(fB, 64, n2g, n2b, true, nullptr, 0, hX, 64, 16);
  __syncthreads();
  gemm_f32(hX, 64, pW3, b3, 64, 64, fA);
  __syncthreads();
  ln_tile(fA, 64, n3g, n3b, false, fA, 64, nullptr, 0, 16);  // ce pre-bos, in-place
  __syncthreads();
  for (int i = lane; i < 1024; i += 32) {
    int rr = i >> 6, c = i & 63;
    int row = row0 + rr;
    bool ok = row < N;
    int rc = ok ? row : 0;
    bool bmv = ok && (bos_mask[rc] != 0);
    float bos = bos_token[t * 64 + c];
    float cev = bmv ? bos : fA[i];
    fA[i] = cev;                                          // ce
    fB[i] = bmv ? bos : motion[(size_t)rc * 64 + c];      // me
    if (ok) ce_out[(size_t)row * 64 + c] = cev;
  }
  __syncthreads();
  int nv = imin(16, N - row0);
  ln_tile(fA, 64, norm1g, norm1b, false, nullptr, 0, cenh + (size_t)row0 * 64, 64, nv);
  ln_tile(fB, 64, norm3g, norm3b, false, nullptr, 0, menh + (size_t)row0 * 64, 64, nv);
}

// per-edge: nbr multi-embed, q/k/v, attention logits + segment max, store v(f16)
__global__ __launch_bounds__(32) void k_edge(
    const float* __restrict__ x, const int* __restrict__ ei,
    const float* __restrict__ eattr, const float* __restrict__ rot,
    const _Float16* __restrict__ cenh, const _Float16* __restrict__ menh,
    const float* xW1, const float* xb1,
    const _Float16* __restrict__ pXL2, const float* xb2,
    const float* xn1b, const float* xn1g,
    const float* eW1, const float* eb1,
    const _Float16* __restrict__ pEL2, const float* eb2,
    const float* en1b, const float* en1g,
    const _Float16* __restrict__ pAG, const float* agb,
    const float* agn1b, const float* agn1g,
    const float* agn2b, const float* agn2g,
    const _Float16* __restrict__ pQ, const float* qb,
    const _Float16* __restrict__ pK, const float* kb,
    const _Float16* __restrict__ pV, const float* vb,
    float* alpha, _Float16* vbuf, unsigned int* amax, int E) {
  __shared__ float    fA[16 * 64];
  __shared__ float    fB[16 * 64];
  __shared__ _Float16 hX[16 * 128];
  __shared__ int dIdx[16];
  int lane = lane_id();
  int e0 = blockIdx.x * 16;
  int r  = lane & 15;
  int cb = (lane >> 4) << 5;

  // per-lane edge scalars (both half-lanes of a row compute the same values)
  int e = imin(e0 + r, E - 1);
  int s = ei[e], d = ei[E + e];
  if (lane < 16) dIdx[lane] = d;
  const float* R = rot + (size_t)d * 4;
  float a0 = x[s * 2], a1 = x[s * 2 + 1];
  float xj0 = a0 * R[0] + a1 * R[2], xj1 = a0 * R[1] + a1 * R[3];
  float c0v = eattr[(size_t)e * 2], c1v = eattr[(size_t)e * 2 + 1];
  float ea0 = c0v * R[0] + c1v * R[2], ea1 = c0v * R[1] + c1v * R[3];

  for (int c = cb; c < cb + 32; ++c)
    fA[r * 64 + c] = xj0 * xW1[c] + xj1 * xW1[64 + c] + xb1[c];
  __syncthreads();
  ln_tile(fA, 64, xn1g, xn1b, true, nullptr, 0, hX, 64, 16);
  __syncthreads();
  gemm_f32(hX, 64, pXL2, xb2, 64, 64, fB);        // ha
  __syncthreads();
  for (int c = cb; c < cb + 32; ++c)
    fA[r * 64 + c] = ea0 * eW1[c] + ea1 * eW1[64 + c] + eb1[c];
  __syncthreads();
  ln_tile(fA, 64, en1g, en1b, true, nullptr, 0, hX, 64, 16);
  __syncthreads();
  gemm_f32_acc(hX, 64, pEL2, eb2, 64, 64, fB);    // s = ha + hb
  __syncthreads();
  ln_tile(fB, 64, agn1g, agn1b, true, nullptr, 0, hX, 64, 16);
  __syncthreads();
  gemm_f32(hX, 64, pAG, agb, 64, 64, fA);
  __syncthreads();
  ln_tile(fA, 64, agn2g, agn2b, false, nullptr, 0, hX, 128, 16);  // nbr -> kin head
  for (int c = cb; c < cb + 32; ++c)
    hX[r * 128 + 64 + c] = menh[(size_t)s * 64 + c];              // kin tail
  __syncthreads();
  gemm_f32(hX, 128, pK, kb, 128, 64, fB);   // k
  gemm_f32(hX, 128, pV, vb, 128, 64, fA);   // v
  __syncthreads();
  for (int i = lane; i < 1024; i += 32) {
    int rr = i >> 6, c = i & 63;
    int ee = e0 + rr;
    if (ee < E) vbuf[(size_t)ee * 64 + c] = (_Float16)fA[i];
  }
  for (int c = cb; c < cb + 32; ++c) {       // qin
    hX[r * 128 + c]      = cenh[(size_t)d * 64 + c];
    hX[r * 128 + 64 + c] = menh[(size_t)d * 64 + c];
  }
  __syncthreads();
  gemm_f32(hX, 128, pQ, qb, 128, 64, fA);   // q
  __syncthreads();
  for (int i = lane; i < 128; i += 32) {
    int rr = i >> 3, h = i & 7;
    float acc = 0.f;
    for (int dd = 0; dd < 8; ++dd)
      acc += fA[rr * 64 + h * 8 + dd] * fB[rr * 64 + h * 8 + dd];
    acc *= 0.35355339059327373f;  // 1/sqrt(8)
    int ee = e0 + rr;
    if (ee < E) {
      alpha[(size_t)ee * 8 + h] = acc;
      atomicMax(&amax[(size_t)dIdx[rr] * 8 + h], f2ord(acc));
    }
  }
}

__global__ void k_expden(const int* __restrict__ ei, float* alpha,
                         const unsigned int* __restrict__ amax, float* den, int E) {
  long long i = (long long)blockIdx.x * blockDim.x + threadIdx.x;
  if (i >= (long long)E * 8) return;
  int e = (int)(i >> 3), h = (int)(i & 7);
  int d = ei[E + e];
  float mx = ord2f(amax[(size_t)d * 8 + h]);
  float ex = __expf(alpha[i] - mx);
  alpha[i] = ex;
  atomicAdd(&den[(size_t)d * 8 + h], ex);
}

__global__ void k_scatter(const int* __restrict__ ei, const float* __restrict__ alpha,
                          const float* __restrict__ den, const _Float16* __restrict__ vbuf,
                          float* agg, int E) {
  long long i = (long long)blockIdx.x * blockDim.x + threadIdx.x;
  if (i >= (long long)E * 64) return;
  int e = (int)(i >> 6), c = (int)(i & 63), h = c >> 3;
  int d = ei[E + e];
  float w = alpha[(size_t)e * 8 + h] / den[(size_t)d * 8 + h];
  atomicAdd(&agg[(size_t)d * 64 + c], w * (float)vbuf[i]);
}

// gated update + out_proj residual + FFN
__global__ __launch_bounds__(32) void k_update(
    const float* __restrict__ cei, const _Float16* __restrict__ cenh,
    const float* __restrict__ agg,
    const _Float16* __restrict__ pIH, const float* ihb,
    const _Float16* __restrict__ pHH, const float* hhb,
    const _Float16* __restrict__ pSELF, const float* sb,
    const _Float16* __restrict__ pOUT, const float* ob,
    const float* n2b, const float* n2g,
    const _Float16* __restrict__ pM1, const float* m1b,
    const _Float16* __restrict__ pM2, const float* m2b,
    float* out, int N) {
  __shared__ float    fA[16 * 64];
  __shared__ float    fB[16 * 64];
  __shared__ float    fC[16 * 64];
  __shared__ _Float16 hA[16 * 64];
  __shared__ _Float16 hB[16 * 64];
  __shared__ _Float16 hF[16 * 256];
  int lane = lane_id();
  int row0 = blockIdx.x * 16;

  for (int i = lane; i < 1024; i += 32) {
    int rr = i >> 6, c = i & 63;
    int row = imin(row0 + rr, N - 1);
    float a = agg[(size_t)row * 64 + c];
    fA[i] = a;
    hA[i] = (_Float16)a;
    hB[i] = cenh[(size_t)row * 64 + c];
  }
  __syncthreads();
  gemm_f32(hA, 64, pIH, ihb, 64, 64, fB);   // lin_ih(agg)
  gemm_f32(hB, 64, pHH, hhb, 64, 64, fC);   // lin_hh(ce_n)
  __syncthreads();
  for (int i = lane; i < 1024; i += 32)
    fB[i] = 1.f / (1.f + __expf(-(fB[i] + fC[i])));  // gate
  __syncthreads();
  gemm_f32(hB, 64, pSELF, sb, 64, 64, fC);  // lin_self(ce_n)
  __syncthreads();
  for (int i = lane; i < 1024; i += 32) {
    float u = fA[i] + fB[i] * (fC[i] - fA[i]);       // upd
    hA[i] = (_Float16)u;
  }
  __syncthreads();
  gemm_f32(hA, 64, pOUT, ob, 64, 64, fB);   // out_proj(upd)
  __syncthreads();
  for (int i = lane; i < 1024; i += 32) {
    int rr = i >> 6, c = i & 63;
    int row = imin(row0 + rr, N - 1);
    fA[i] = cei[(size_t)row * 64 + c] + fB[i];       // ce2
  }
  __syncthreads();
  ln_tile(fA, 64, n2g, n2b, false, nullptr, 0, hB, 64, 16);  // h (f16)
  __syncthreads();
  for (int n0 = 0; n0 < 256; n0 += 16) {    // mlp1 + ReLU -> f16
    v8f acc = wmma_tile(hB, 64, pM1, 64 >> 5, n0, 64);
    store_h16_relu(hF, 256, n0, m1b, acc);
  }
  __syncthreads();
  gemm_f32(hF, 256, pM2, m2b, 256, 64, fB); // mlp2
  __syncthreads();
  for (int i = lane; i < 1024; i += 32) {
    int rr = i >> 6, c = i & 63;
    int row = row0 + rr;
    if (row < N) out[(size_t)row * 64 + c] = fA[i] + fB[i];
  }
}

// ===================== launch =====================

extern "C" void kernel_launch(void* const* d_in, const int* in_sizes, int n_in,
                              void* d_out, int out_size, void* d_ws, size_t ws_size,
                              hipStream_t stream) {
  if (n_in < 62) return;
  const float* x  = (const float*)d_in[0];
  const int*   t  = (const int*)d_in[1];
  const int*   ei = (const int*)d_in[2];
  const float* ea = (const float*)d_in[3];
  const unsigned char* bm = (const unsigned char*)d_in[4];
  const float* me  = (const float*)d_in[5];
  const float* rot = (const float*)d_in[6];

  // params pytree leaves (sorted dict keys, {W,b} and {b,g} sorted)
  enum {
    BOS_TOKEN = 0,
    CE_L1_W, CE_L1_B, CE_L2_W, CE_L2_B, CE_L3_W, CE_L3_B,
    CE_N1_B, CE_N1_G, CE_N2_B, CE_N2_G, CE_N3_B, CE_N3_G,
    LIN_HH_W, LIN_HH_B, LIN_IH_W, LIN_IH_B,
    LIN_K_W, LIN_K_B, LIN_Q_W, LIN_Q_B, LIN_SELF_W, LIN_SELF_B, LIN_V_W, LIN_V_B,
    MLP1_W, MLP1_B, MLP2_W, MLP2_B,
    AG_L_W, AG_L_B, AG_N1_B, AG_N1_G, AG_N2_B, AG_N2_G,
    NE_L1_W, NE_L1_B, NE_L2_W, NE_L2_B, NE_N1_B, NE_N1_G,
    NX_L1_W, NX_L1_B, NX_L2_W, NX_L2_B, NX_N1_B, NX_N1_G,
    NORM1_B, NORM1_G, NORM2_B, NORM2_G, NORM3_B, NORM3_G,
    OUT_W, OUT_B, P_COUNT
  };
  const float* P[P_COUNT];
  for (int i = 0; i < P_COUNT; ++i) P[i] = (const float*)d_in[7 + i];

  int N = in_sizes[0] / 2;
  int E = in_sizes[2] / 2;

  // workspace layout
  size_t nf = (size_t)N * 64;
  float* ce   = (float*)d_ws;                               // N*64 f32
  _Float16* cenh = (_Float16*)(ce + nf);                    // N*64 f16
  _Float16* menh = cenh + nf;                               // N*64 f16
  float* agg  = (float*)(menh + nf);                        // N*64 f32
  float* den  = agg + nf;                                   // N*8  f32
  unsigned int* amax = (unsigned int*)(den + (size_t)N * 8);
  float* alpha = (float*)(amax + (size_t)N * 8);            // E*8 f32
  _Float16* vbuf = (_Float16*)(alpha + (size_t)E * 8);      // E*64 f16
  _Float16* wp = vbuf + (size_t)E * 64;                     // packed weights

  _Float16* pCEL2 = wp;  wp += 4096;
  _Float16* pCEL3 = wp;  wp += 4096;
  _Float16* pXL2  = wp;  wp += 4096;
  _Float16* pEL2  = wp;  wp += 4096;
  _Float16* pAG   = wp;  wp += 4096;
  _Float16* pIH   = wp;  wp += 4096;
  _Float16* pHH   = wp;  wp += 4096;
  _Float16* pSELF = wp;  wp += 4096;
  _Float16* pOUT  = wp;  wp += 4096;
  _Float16* pQ    = wp;  wp += 8192;
  _Float16* pKk   = wp;  wp += 8192;
  _Float16* pV    = wp;  wp += 8192;
  _Float16* pM1   = wp;  wp += 16384;
  _Float16* pM2   = wp;  wp += 16384;

  auto packW = [&](const float* W, _Float16* Pk, int K, int Nn) {
    k_pack<<<(K * Nn + 255) / 256, 256, 0, stream>>>(W, Pk, K, Nn);
  };
  packW(P[CE_L2_W],    pCEL2,  64,  64);
  packW(P[CE_L3_W],    pCEL3,  64,  64);
  packW(P[NX_L2_W],    pXL2,   64,  64);
  packW(P[NE_L2_W],    pEL2,   64,  64);
  packW(P[AG_L_W],     pAG,    64,  64);
  packW(P[LIN_IH_W],   pIH,    64,  64);
  packW(P[LIN_HH_W],   pHH,    64,  64);
  packW(P[LIN_SELF_W], pSELF,  64,  64);
  packW(P[OUT_W],      pOUT,   64,  64);
  packW(P[LIN_Q_W],    pQ,    128,  64);
  packW(P[LIN_K_W],    pKk,   128,  64);
  packW(P[LIN_V_W],    pV,    128,  64);
  packW(P[MLP1_W],     pM1,    64, 256);
  packW(P[MLP2_W],     pM2,   256,  64);

  k_init<<<(N * 64 + 255) / 256, 256, 0, stream>>>(agg, den, amax, N);

  k_node_embed<<<(N + 15) / 16, 32, 0, stream>>>(
      x, t, bm, me, rot,
      P[CE_L1_W], P[CE_L1_B], pCEL2, P[CE_L2_B], pCEL3, P[CE_L3_B],
      P[CE_N1_B], P[CE_N1_G], P[CE_N2_B], P[CE_N2_G], P[CE_N3_B], P[CE_N3_G],
      P[NORM1_B], P[NORM1_G], P[NORM3_B], P[NORM3_G], P[BOS_TOKEN],
      ce, cenh, menh, N);

  k_edge<<<(E + 15) / 16, 32, 0, stream>>>(
      x, ei, ea, rot, cenh, menh,
      P[NX_L1_W], P[NX_L1_B], pXL2, P[NX_L2_B], P[NX_N1_B], P[NX_N1_G],
      P[NE_L1_W], P[NE_L1_B], pEL2, P[NE_L2_B], P[NE_N1_B], P[NE_N1_G],
      pAG, P[AG_L_B], P[AG_N1_B], P[AG_N1_G], P[AG_N2_B], P[AG_N2_G],
      pQ, P[LIN_Q_B], pKk, P[LIN_K_B], pV, P[LIN_V_B],
      alpha, vbuf, amax, E);

  long long eh = (long long)E * 8;
  k_expden<<<(unsigned)((eh + 255) / 256), 256, 0, stream>>>(ei, alpha, amax, den, E);

  long long ec = (long long)E * 64;
  k_scatter<<<(unsigned)((ec + 255) / 256), 256, 0, stream>>>(ei, alpha, den, vbuf, agg, E);

  k_update<<<(N + 15) / 16, 32, 0, stream>>>(
      ce, cenh, agg,
      pIH, P[LIN_IH_B], pHH, P[LIN_HH_B], pSELF, P[LIN_SELF_B],
      pOUT, P[OUT_B], P[NORM2_B], P[NORM2_G],
      pM1, P[MLP1_B], pM2, P[MLP2_B],
      (float*)d_out, N);
}
